// ReasonEncoder_21311627722946
// MI455X (gfx1250) — compile-verified, matrix operand order checked by reading, not run on hardware
//
#include <hip/hip_runtime.h>

#define EMBED 256
#define UNITS 256
#define GATES 1024
#define BSZ   256
#define TLEN  128

typedef __attribute__((ext_vector_type(16))) __bf16 v16bf;
typedef __attribute__((ext_vector_type(8)))  __bf16 v8bf;
typedef __attribute__((ext_vector_type(8)))  float  v8f;
typedef __attribute__((ext_vector_type(8)))  unsigned short usv8;

// round-to-nearest-even f32 -> bf16 bits (bit-level, toolchain-portable)
static __device__ __forceinline__ unsigned short bf16_bits(float f) {
    union { float f; unsigned u; } x; x.f = f;
    unsigned r = x.u + 0x7FFFu + ((x.u >> 16) & 1u);
    return (unsigned short)(r >> 16);
}

// 16-bit A-matrix 16x32 layout: lane l holds M = l%16; element pair jj holds
// K = base(jj) + {0,1} + 8*(l>=16); base: jj<4 -> 2jj, jj>=4 -> 16+2(jj-4).
// => per lane the 16 elements are two contiguous 8-element runs:
//    k0+8*hi+[0,8) and k0+16+8*hi+[0,8)
static __device__ __forceinline__ int kbase(int jj) {
    return (jj < 4) ? (2 * jj) : (16 + 2 * (jj - 4));
}

static __device__ __forceinline__ v8f wmma_bf16(v16bf a, v16bf b, v8f c) {
    return __builtin_amdgcn_wmma_f32_16x16x32_bf16(
        false, a, false, b, (short)0, c, false, false);
}

// Build A-fragment from row-major bf16 LDS tile [16][256]: 2x ds_load_b128
static __device__ __forceinline__ v16bf load_afrag(const unsigned short* lds,
                                                   int rowbase, int k0) {
    const unsigned short* p = lds + rowbase + k0;
    v8bf lo = *(const v8bf*)(const void*)(p);
    v8bf hi = *(const v8bf*)(const void*)(p + 16);
    return __builtin_shufflevector(lo, hi, 0, 1, 2, 3, 4, 5, 6, 7,
                                   8, 9, 10, 11, 12, 13, 14, 15);
}

// ---------------------------------------------------------------------------
// Kernel 0: pre-swizzle W/U (f32 [256][1024] row-major) into bf16 WMMA
// B-fragments: element (lane, j) = src[32*kt + kbase(j/2) + (j&1) +
// 8*(lane>=16)][16*nt + lane%16], contiguous per lane (one 32B load).
// dst index e = ((((mat*2+dir)*64 + nt)*8) + kt)*512 + lane*16 + j
// ---------------------------------------------------------------------------
__global__ __launch_bounds__(256)
void swizzle_kernel(const float* __restrict__ Wf, const float* __restrict__ Uf,
                    const float* __restrict__ Wb, const float* __restrict__ Ub,
                    unsigned short* __restrict__ frag) {
    int e = blockIdx.x * blockDim.x + threadIdx.x;
    const int TOT = 2 * 2 * 64 * 8 * 512;
    if (e >= TOT) return;
    int within = e & 511;
    int kt  = (e >> 9)  & 7;
    int nt  = (e >> 12) & 63;
    int dir = (e >> 18) & 1;
    int mat = (e >> 19) & 1;
    int lane = within >> 4;
    int j    = within & 15;
    int k = 32 * kt + kbase(j >> 1) + (j & 1) + ((lane >= 16) ? 8 : 0);
    int n = 16 * nt + (lane & 15);
    const float* src = (mat == 0) ? (dir ? Ub : Uf) : (dir ? Wb : Wf);
    frag[e] = bf16_bits(src[k * GATES + n]);
}

// ---------------------------------------------------------------------------
// Kernel 1: recurrence. Grid (16 batch-chunks, 2 directions); blocks fully
// independent (LSTM recurrence is per-batch-row). 16 waves; wave w owns
// hidden cols [16w,16w+16) and its 4 gate tiles. c in C-fragment registers;
// h kept in LDS twice: f32 (state carry) + bf16 (WMMA A operand).
// ---------------------------------------------------------------------------
__global__ __launch_bounds__(512)
void lstm_kernel(const int* __restrict__ x, const float* __restrict__ emb_table,
                 const float* __restrict__ bf_, const float* __restrict__ bb_,
                 const unsigned short* __restrict__ frag, float* __restrict__ out) {
    __shared__ float          hbuf[16 * 256];     // f32 state
    __shared__ unsigned short hbf[16 * 256];      // bf16 copy for WMMA
    __shared__ unsigned short ebf[16 * 256];      // bf16 embeddings
    __shared__ int idxbuf[16];
    __shared__ unsigned mbbuf;

    const int tid  = threadIdx.x;
    const int lane = tid & 31;
    const int w    = tid >> 5;
    const int d    = blockIdx.y;
    const int b0   = blockIdx.x * 16;
    const int hi   = (lane >> 4) & 1;
    const int rowbase = (lane & 15) * 256 + 8 * hi;   // A-frag per-lane base

    const float* bias = d ? bb_ : bf_;
    const unsigned short* Ufrag = frag + d * (64 * 8 * 512);
    const unsigned short* Wfrag = frag + 2 * (64 * 8 * 512) + d * (64 * 8 * 512);

    const int n_h = 16 * w + (lane & 15);
    const float bI = bias[n_h];
    const float bF = bias[256 + n_h];
    const float bG = bias[512 + n_h];
    const float bO = bias[768 + n_h];

    for (int i = tid; i < 16 * 256; i += 512) { hbuf[i] = 0.f; hbf[i] = 0; }
    float c_reg[8];
#pragma unroll
    for (int v = 0; v < 8; ++v) c_reg[v] = 0.f;
    __syncthreads();

    for (int t = 0; t < TLEN; ++t) {
        // Defeat LICM of the weight-fragment loads: without this the compiler
        // hoists 64 fragments (512 VGPRs worth) out of the loop and spills
        // them all to scratch. With it they stream from (L2-resident) global.
        asm volatile("" ::: "memory");

        const int tt = d ? (TLEN - 1 - t) : t;

        // phase 1: token indices
        if (tid < 16) idxbuf[tid] = x[(b0 + tid) * TLEN + tt];
        __syncthreads();

        if (tid == 0) {
            unsigned mbv = 0;
#pragma unroll
            for (int r = 0; r < 16; ++r) mbv |= (idxbuf[r] != 0) ? (1u << r) : 0u;
            mbbuf = mbv;
        }
        // phase 2: gather embeddings, convert to bf16 (8 elems/thread)
        {
            int r  = tid >> 5;
            int c0 = (tid & 31) * 8;
            const float* src = emb_table + idxbuf[r] * EMBED + c0;
            float4 a = *(const float4*)(src);
            float4 b = *(const float4*)(src + 4);
            usv8 u;
            u[0] = bf16_bits(a.x); u[1] = bf16_bits(a.y);
            u[2] = bf16_bits(a.z); u[3] = bf16_bits(a.w);
            u[4] = bf16_bits(b.x); u[5] = bf16_bits(b.y);
            u[6] = bf16_bits(b.z); u[7] = bf16_bits(b.w);
            *(usv8*)(void*)(ebf + r * 256 + c0) = u;
        }
        __syncthreads();
        const unsigned mb = mbbuf;

        // z = h @ U + e @ W, f32 accumulate, per-gate tiles
        v8f accI = {}, accF = {}, accG = {}, accO = {};
#pragma unroll
        for (int kt = 0; kt < 8; ++kt) {
            v16bf a = load_afrag(hbf, rowbase, kt * 32);
            v16bf b0 = *(const v16bf*)(const void*)(Ufrag + ((w     ) * 8 + kt) * 512 + lane * 16);
            v16bf b1 = *(const v16bf*)(const void*)(Ufrag + ((w + 16) * 8 + kt) * 512 + lane * 16);
            v16bf b2 = *(const v16bf*)(const void*)(Ufrag + ((w + 32) * 8 + kt) * 512 + lane * 16);
            v16bf b3 = *(const v16bf*)(const void*)(Ufrag + ((w + 48) * 8 + kt) * 512 + lane * 16);
            accI = wmma_bf16(a, b0, accI);
            accF = wmma_bf16(a, b1, accF);
            accG = wmma_bf16(a, b2, accG);
            accO = wmma_bf16(a, b3, accO);
        }
#pragma unroll
        for (int kt = 0; kt < 8; ++kt) {
            v16bf a = load_afrag(ebf, rowbase, kt * 32);
            v16bf b0 = *(const v16bf*)(const void*)(Wfrag + ((w     ) * 8 + kt) * 512 + lane * 16);
            v16bf b1 = *(const v16bf*)(const void*)(Wfrag + ((w + 16) * 8 + kt) * 512 + lane * 16);
            v16bf b2 = *(const v16bf*)(const void*)(Wfrag + ((w + 32) * 8 + kt) * 512 + lane * 16);
            v16bf b3 = *(const v16bf*)(const void*)(Wfrag + ((w + 48) * 8 + kt) * 512 + lane * 16);
            accI = wmma_bf16(a, b0, accI);
            accF = wmma_bf16(a, b1, accF);
            accG = wmma_bf16(a, b2, accG);
            accO = wmma_bf16(a, b3, accO);
        }

        __syncthreads();   // all waves done reading hbf for this step

        // gates + state; C-fragment: VGPR v -> M = v + 8*hi, N = lane%16
#pragma unroll
        for (int v = 0; v < 8; ++v) {
            int m = v + 8 * hi;
            float zi = accI[v] + bI;
            float zf = accF[v] + bF;
            float zg = accG[v] + bG;
            float zo = accO[v] + bO;
            float ig = 1.f / (1.f + __expf(-zi));
            float fg = 1.f / (1.f + __expf(-zf));
            float gg = tanhf(zg);
            float og = 1.f / (1.f + __expf(-zo));
            float cn = fg * c_reg[v] + ig * gg;
            float hn = og * tanhf(cn);
            bool keep = (mb >> m) & 1u;
            c_reg[v] = keep ? cn : c_reg[v];
            int addr = m * 256 + n_h;             // lane-private slot
            float hv = keep ? hn : hbuf[addr];
            hbuf[addr] = hv;
            hbf[addr]  = bf16_bits(hv);
        }
        // writes ordered before next step's reads by the two barriers above
    }
    __syncthreads();

    for (int i = tid; i < 16 * 256; i += 512) {
        int r = i >> 8, c = i & 255;
        out[(b0 + r) * 512 + d * 256 + c] = hbuf[i];
    }
}

extern "C" void kernel_launch(void* const* d_in, const int* in_sizes, int n_in,
                              void* d_out, int out_size, void* d_ws, size_t ws_size,
                              hipStream_t stream) {
    const int*   x   = (const int*)  d_in[0];
    const float* emb = (const float*)d_in[1];
    const float* Wf  = (const float*)d_in[2];
    const float* Uf  = (const float*)d_in[3];
    const float* bf_ = (const float*)d_in[4];
    const float* Wb  = (const float*)d_in[5];
    const float* Ub  = (const float*)d_in[6];
    const float* bb_ = (const float*)d_in[7];
    float* out = (float*)d_out;
    unsigned short* frag = (unsigned short*)d_ws;   // 2 MB bf16 fragments

    const int TOT = 2 * 2 * 64 * 8 * 512;
    swizzle_kernel<<<TOT / 256, 256, 0, stream>>>(Wf, Uf, Wb, Ub, frag);
    lstm_kernel<<<dim3(16, 2), 512, 0, stream>>>(x, emb, bf_, bb_, frag, out);
}
